// MHAttention_71554155151928
// MI455X (gfx1250) — matrix operand order repaired and checked by reading, not compiled, not run on hardware
//
#include <hip/hip_runtime.h>

#define EMB 1024
#define HEADS 16
#define HD 64
#define SEQ 2048

typedef __attribute__((ext_vector_type(16))) _Float16 v16h;
typedef __attribute__((ext_vector_type(8)))  _Float16 v8h;
typedef __attribute__((ext_vector_type(8)))  float    v8f;
typedef __attribute__((ext_vector_type(4)))  int      v4i;

#if __has_builtin(__builtin_amdgcn_global_load_async_to_lds_b128)
#define ASYNC_LDS 1
typedef __attribute__((address_space(1))) v4i glob_v4i;  // global <4 x i32>
typedef __attribute__((address_space(3))) v4i lds_v4i;   // LDS    <4 x i32>
#else
#define ASYNC_LDS 0
#endif

__device__ __forceinline__ v16h cat8(v8h lo, v8h hi) {
  return __builtin_shufflevector(lo, hi, 0,1,2,3,4,5,6,7,8,9,10,11,12,13,14,15);
}

// A-matrix fragment (16x32 f16): lane holds row m = lane%16.
// VGPR0..3 <- K = half*8 + 0..7 ; VGPR4..7 <- K = 16 + half*8 + 0..7 (ISA 7.12.2)
__device__ __forceinline__ v16h load_a_f16(const _Float16* base, int ld, int lane) {
  int m = lane & 15, hf = lane >> 4;
  const _Float16* p = base + (size_t)m * ld + hf * 8;
  v8h lo = *(const v8h*)p;
  v8h hi = *(const v8h*)(p + 16);
  return cat8(lo, hi);
}

__device__ __forceinline__ v16h load_a_f32(const float* base, int ld, int lane) {
  int m = lane & 15, hf = lane >> 4;
  const float* p0 = base + (size_t)m * ld + hf * 8;
  const float* p1 = p0 + 16;
  v16h r;
#pragma unroll
  for (int i = 0; i < 8; ++i) { r[i] = (_Float16)p0[i]; r[i + 8] = (_Float16)p1[i]; }
  return r;
}

// B-matrix fragment (32x16 f16): lane holds column n = lane%16,
// K = half*8 + 0..7, then 16 + half*8 + 0..7. colbase: column n's data at colbase + n*ld.
__device__ __forceinline__ v16h load_b_f16(const _Float16* colbase, int ld, int lane) {
  int n = lane & 15, hf = lane >> 4;
  const _Float16* p = colbase + (size_t)n * ld + hf * 8;
  v8h lo = *(const v8h*)p;
  v8h hi = *(const v8h*)(p + 16);
  return cat8(lo, hi);
}

__device__ __forceinline__ v16h load_b_f32(const float* colbase, int ld, int lane) {
  int n = lane & 15, hf = lane >> 4;
  const float* p = colbase + (size_t)n * ld + hf * 8;
  v16h r;
#pragma unroll
  for (int i = 0; i < 8; ++i) { r[i] = (_Float16)p[i]; r[i + 8] = (_Float16)p[i + 16]; }
  return r;
}

__device__ __forceinline__ v8f wmma16(v16h a, v16h b, v8f c) {
  // D = A(16x32) x B(32x16) + C, f32 accumulate
  return __builtin_amdgcn_wmma_f32_16x16x32_f16(false, a, false, b, (short)0, c, false, false);
}

// 16-byte global -> LDS copy: async DMA (ASYNCcnt) on CDNA5 when available,
// else synchronous VGPR round-trip.
__device__ __forceinline__ void cp16(_Float16* l, const _Float16* g) {
#if ASYNC_LDS
  __builtin_amdgcn_global_load_async_to_lds_b128((glob_v4i*)g, (lds_v4i*)l, 0, 0);
#else
  *(v8h*)l = *(const v8h*)g;
#endif
}

__device__ __forceinline__ void wait_async() {
#if ASYNC_LDS
  __asm__ volatile("s_wait_asynccnt 0" ::: "memory");
#endif
}

// ---------------------------------------------------------------------------
// Kernel 1: per-head Q/K/V projections.  y = x @ W^T + b  (W:[HD,HD] row-major,
// so B-matrix column e is W's row e -> contiguous).  Q pre-scaled by 1/sqrt(HD).
// Q,K stored [bh][s][d] f16;  V stored transposed [bh][d][s] f16.
// ---------------------------------------------------------------------------
__device__ __forceinline__ void proj_one(v16h a0, v16h a1, const float* W,
                                         const float* bias, float scale,
                                         _Float16* out, long ss, long se, int lane) {
  int nn = lane & 15, hf = lane >> 4;
#pragma unroll
  for (int n = 0; n < 4; ++n) {
    v8f c = {};
    v16h b0 = load_b_f32(W + n * 16 * HD, HD, lane);       // k-chunk 0..31
    c = wmma16(a0, b0, c);
    v16h b1 = load_b_f32(W + n * 16 * HD + 32, HD, lane);  // k-chunk 32..63
    c = wmma16(a1, b1, c);
    float bb = bias[n * 16 + nn];
#pragma unroll
    for (int r = 0; r < 8; ++r) {
      float v = (c[r] + bb) * scale;
      out[(long)(r + 8 * hf) * ss + (long)(n * 16 + nn) * se] = (_Float16)v;
    }
  }
}

__global__ __launch_bounds__(256) void proj_kernel(
    const float* __restrict__ X, const float* __restrict__ Z,
    const float* __restrict__ Wq, const float* __restrict__ bq,
    const float* __restrict__ Wk, const float* __restrict__ bk,
    const float* __restrict__ Wv, const float* __restrict__ bv,
    _Float16* __restrict__ Qh, _Float16* __restrict__ Kh, _Float16* __restrict__ Vt) {
  int bh = blockIdx.x, b = bh / HEADS, h = bh % HEADS;
  int wave = threadIdx.x >> 5, lane = threadIdx.x & 31;
  int s0 = blockIdx.y * 128 + wave * 16;
  const float* Xr = X + ((size_t)b * SEQ + s0) * EMB + h * HD;
  const float* Zr = Z + ((size_t)b * SEQ + s0) * EMB + h * HD;
  v16h ax0 = load_a_f32(Xr, EMB, lane), ax1 = load_a_f32(Xr + 32, EMB, lane);
  v16h az0 = load_a_f32(Zr, EMB, lane), az1 = load_a_f32(Zr + 32, EMB, lane);
  proj_one(ax0, ax1, Wq + (size_t)h * HD * HD, bq + h * HD, 0.125f,
           Qh + ((size_t)bh * SEQ + s0) * HD, HD, 1, lane);
  proj_one(az0, az1, Wk + (size_t)h * HD * HD, bk + h * HD, 1.0f,
           Kh + ((size_t)bh * SEQ + s0) * HD, HD, 1, lane);
  proj_one(az0, az1, Wv + (size_t)h * HD * HD, bv + h * HD, 1.0f,
           Vt + (size_t)bh * HD * SEQ + s0, 1, SEQ, lane);
}

// ---------------------------------------------------------------------------
// Kernel 2: causal flash attention.  Block = 8 waves x 16 q-rows = 128 rows of
// one (b,h).  64-key blocks, double-buffered via async global->LDS copies.
// Online softmax state (m,l) lives per-lane, aligned with C-fragment rows.
// ---------------------------------------------------------------------------
__global__ __launch_bounds__(256) void attn_kernel(
    const _Float16* __restrict__ Qh, const _Float16* __restrict__ Kh,
    const _Float16* __restrict__ Vt, _Float16* __restrict__ O) {
  int bh = blockIdx.x, b = bh / HEADS, h = bh % HEADS;
  int wave = threadIdx.x >> 5, lane = threadIdx.x & 31;
  int nn = lane & 15, hf = lane >> 4;
  int qb0 = blockIdx.y * 128;
  int qw = qb0 + wave * 16;
  const _Float16* Qb = Qh + ((size_t)bh * SEQ + qw) * HD;
  const _Float16* Kb = Kh + (size_t)bh * SEQ * HD;
  const _Float16* Vb = Vt + (size_t)bh * HD * SEQ;

  __shared__ alignas(16) _Float16 Kt[2][64 * HD];    // [buf][key][d]   8KB/buf
  __shared__ alignas(16) _Float16 Vts[2][HD * 64];   // [buf][d][key]   8KB/buf
  __shared__ alignas(16) _Float16 Pst[8][16 * 64];   // per-wave P staging

  v16h qa0 = load_a_f16(Qb, HD, lane);
  v16h qa1 = load_a_f16(Qb + 32, HD, lane);
  v8f acc[4] = {};
  float mrow[8], lrow[8];
#pragma unroll
  for (int r = 0; r < 8; ++r) { mrow[r] = -1e30f; lrow[r] = 0.0f; }

  // cooperative 64x64 K tile + 64x64 V^T tile stage (16 f16 per thread each)
  auto stage = [&](int buf, int k0) {
    int t = (int)threadIdx.x;
    int kr = t >> 2, cc = (t & 3) * 16;
    cp16(&Kt[buf][kr * HD + cc],      Kb + (size_t)(k0 + kr) * HD + cc);
    cp16(&Kt[buf][kr * HD + cc + 8],  Kb + (size_t)(k0 + kr) * HD + cc + 8);
    cp16(&Vts[buf][kr * 64 + cc],     Vb + (size_t)kr * SEQ + k0 + cc);
    cp16(&Vts[buf][kr * 64 + cc + 8], Vb + (size_t)kr * SEQ + k0 + cc + 8);
    // pull tile-after-next toward L2 (speculative; ws region stays mapped)
    __builtin_prefetch(Kb + (size_t)(k0 + 64 + kr) * HD + cc, 0, 0);
    __builtin_prefetch(Vb + (size_t)kr * SEQ + k0 + 64 + cc, 0, 0);
  };

  int nkb = qb0 / 64 + 2;  // causal: 64-key blocks covering [0, qb0+127]
  stage(0, 0);
  for (int kb = 0; kb < nkb; ++kb) {
    int cur = kb & 1;
    int k0 = kb * 64;
    wait_async();       // my wave's async copies (incl. into `cur`) landed
    __syncthreads();    // every wave's copies landed; prev reads of `nxt` done
    if (kb + 1 < nkb) stage(1 - cur, k0 + 64);

    const _Float16* Kc = &Kt[cur][0];
    const _Float16* Vc = &Vts[cur][0];

    // S = Q K^T : four 16x16 fragments over keys [k0+16j, k0+16j+15]
    v8f s[4] = {};
#pragma unroll
    for (int nf = 0; nf < 4; ++nf) {
      s[nf] = wmma16(qa0, load_b_f16(Kc + nf * 16 * HD, HD, lane), s[nf]);
      s[nf] = wmma16(qa1, load_b_f16(Kc + nf * 16 * HD + 32, HD, lane), s[nf]);
    }

    _Float16* Pw = &Pst[wave][0];
#pragma unroll
    for (int r = 0; r < 8; ++r) {
      int qi = qw + r + 8 * hf;
      float a0 = s[0][r]; if (k0 + nn > qi)      a0 = -1e30f;  // causal mask
      float a1 = s[1][r]; if (k0 + 16 + nn > qi) a1 = -1e30f;
      float a2 = s[2][r]; if (k0 + 32 + nn > qi) a2 = -1e30f;
      float a3 = s[3][r]; if (k0 + 48 + nn > qi) a3 = -1e30f;
      float mx = fmaxf(fmaxf(a0, a1), fmaxf(a2, a3));
      mx = fmaxf(mx, __shfl_xor(mx, 1, 16));
      mx = fmaxf(mx, __shfl_xor(mx, 2, 16));
      mx = fmaxf(mx, __shfl_xor(mx, 4, 16));
      mx = fmaxf(mx, __shfl_xor(mx, 8, 16));
      float mnew = fmaxf(mrow[r], mx);
      float sc = __expf(mrow[r] - mnew);
      float p0 = __expf(a0 - mnew), p1 = __expf(a1 - mnew);
      float p2 = __expf(a2 - mnew), p3 = __expf(a3 - mnew);
      float rs = (p0 + p1) + (p2 + p3);
      rs += __shfl_xor(rs, 1, 16);
      rs += __shfl_xor(rs, 2, 16);
      rs += __shfl_xor(rs, 4, 16);
      rs += __shfl_xor(rs, 8, 16);
      lrow[r] = lrow[r] * sc + rs;
      mrow[r] = mnew;
#pragma unroll
      for (int n = 0; n < 4; ++n) acc[n][r] *= sc;
      int rowo = (r + 8 * hf) * 64;  // C-layout -> LDS row-major 16x64
      Pw[rowo + nn]      = (_Float16)p0;
      Pw[rowo + 16 + nn] = (_Float16)p1;
      Pw[rowo + 32 + nn] = (_Float16)p2;
      Pw[rowo + 48 + nn] = (_Float16)p3;
    }
    __asm__ volatile("s_wait_dscnt 0" ::: "memory");  // wave-local LDS RAW
    v16h pa0 = load_a_f16(Pw, 64, lane);       // keys k0   .. k0+31
    v16h pa1 = load_a_f16(Pw + 32, 64, lane);  // keys k0+32.. k0+63
#pragma unroll
    for (int n = 0; n < 4; ++n) {
      acc[n] = wmma16(pa0, load_b_f16(Vc + n * 16 * 64, 64, lane), acc[n]);
      acc[n] = wmma16(pa1, load_b_f16(Vc + n * 16 * 64 + 32, 64, lane), acc[n]);
    }
  }

  _Float16* Ob = O + (size_t)b * SEQ * EMB + h * HD;
#pragma unroll
  for (int r = 0; r < 8; ++r) {
    float inv = 1.0f / lrow[r];
    size_t row = (size_t)(qw + r + 8 * hf) * EMB;
    Ob[row + nn]      = (_Float16)(acc[0][r] * inv);
    Ob[row + 16 + nn] = (_Float16)(acc[1][r] * inv);
    Ob[row + 32 + nn] = (_Float16)(acc[2][r] * inv);
    Ob[row + 48 + nn] = (_Float16)(acc[3][r] * inv);
  }
}

// ---------------------------------------------------------------------------
// Kernel 3: Y = O @ Wo^T + bo (f32 out).  Wo tile (64 e x 32 c) converted to
// f16 and staged in LDS per k-step; 8 waves x 16 rows per block.
// ---------------------------------------------------------------------------
__global__ __launch_bounds__(256) void oproj_kernel(
    const _Float16* __restrict__ O, const float* __restrict__ Wo,
    const float* __restrict__ bo, float* __restrict__ Y) {
  int wave = threadIdx.x >> 5, lane = threadIdx.x & 31;
  int nn = lane & 15, hf = lane >> 4;
  int r0 = blockIdx.x * 128 + wave * 16;
  int e0 = blockIdx.y * 64;
  __shared__ alignas(16) _Float16 Wt[64 * 32];  // [e_local][c_local]
  v8f c0 = {}, c1 = {}, c2 = {}, c3 = {};
  for (int cb = 0; cb < EMB / 32; ++cb) {
    __syncthreads();
    {  // stage + f32->f16 convert: 8 contiguous elems per thread
      int el = threadIdx.x >> 2, cc = (threadIdx.x & 3) * 8;
      const float* src = Wo + (size_t)(e0 + el) * EMB + cb * 32 + cc;
      v8h t;
#pragma unroll
      for (int i = 0; i < 8; ++i) t[i] = (_Float16)src[i];
      *(v8h*)&Wt[el * 32 + cc] = t;
    }
    __syncthreads();
    v16h a = load_a_f16(O + (size_t)r0 * EMB + cb * 32, EMB, lane);
    c0 = wmma16(a, load_b_f16(&Wt[0],       32, lane), c0);
    c1 = wmma16(a, load_b_f16(&Wt[16 * 32], 32, lane), c1);
    c2 = wmma16(a, load_b_f16(&Wt[32 * 32], 32, lane), c2);
    c3 = wmma16(a, load_b_f16(&Wt[48 * 32], 32, lane), c3);
  }
#pragma unroll
  for (int r = 0; r < 8; ++r) {
    size_t row = (size_t)(r0 + r + 8 * hf) * EMB + e0;
    Y[row + nn]      = c0[r] + bo[e0 + nn];
    Y[row + 16 + nn] = c1[r] + bo[e0 + 16 + nn];
    Y[row + 32 + nn] = c2[r] + bo[e0 + 32 + nn];
    Y[row + 48 + nn] = c3[r] + bo[e0 + 48 + nn];
  }
}

extern "C" void kernel_launch(void* const* d_in, const int* in_sizes, int n_in,
                              void* d_out, int out_size, void* d_ws, size_t ws_size,
                              hipStream_t stream) {
  const float* X  = (const float*)d_in[0];
  const float* Z  = (const float*)d_in[1];
  const float* Wq = (const float*)d_in[2];
  const float* bq = (const float*)d_in[3];
  const float* Wk = (const float*)d_in[4];
  const float* bk = (const float*)d_in[5];
  const float* Wv = (const float*)d_in[6];
  const float* bv = (const float*)d_in[7];
  const float* Wo = (const float*)d_in[8];
  const float* bo = (const float*)d_in[9];
  int B = in_sizes[0] / (SEQ * EMB);

  size_t per = (size_t)B * HEADS * SEQ * HD;  // f16 element count per tensor
  _Float16* Qh = (_Float16*)d_ws;
  _Float16* Kh = Qh + per;
  _Float16* Vt = Kh + per;
  _Float16* O  = Vt + per;
  float* Y = (float*)d_out;

  dim3 blk(256);
  dim3 g1(B * HEADS, SEQ / 128);
  proj_kernel<<<g1, blk, 0, stream>>>(X, Z, Wq, bq, Wk, bk, Wv, bv, Qh, Kh, Vt);
  attn_kernel<<<g1, blk, 0, stream>>>(Qh, Kh, Vt, O);
  dim3 g3((B * SEQ) / 128, EMB / 64);
  oproj_kernel<<<g3, blk, 0, stream>>>(O, Wo, bo, Y);
}
